// BackgroundEmbeddingTransformerLayer_26980984553695
// MI455X (gfx1250) — matrix operand order repaired and checked by reading, not compiled, not run on hardware
//
#include <hip/hip_runtime.h>

// ---------------------------------------------------------------------------
// Types
// ---------------------------------------------------------------------------
typedef __attribute__((ext_vector_type(16))) __bf16       v16bf;
typedef __attribute__((ext_vector_type(8)))  float        v8f;
typedef __attribute__((ext_vector_type(4)))  float        f32x4;
typedef __attribute__((ext_vector_type(4)))  unsigned int u32x4;
typedef __attribute__((ext_vector_type(8)))  int          i32x8;
typedef __attribute__((ext_vector_type(4)))  int          i32x4;

// ---------------------------------------------------------------------------
// Problem constants (fixed by the reference)
// ---------------------------------------------------------------------------
#define BSZ   8
#define NLVL  4
#define EDIM  512
#define NH    8
#define DHEAD 64
#define FFND  2048
#define NNZ   131072
#define LMAX  16384
#define SPLIT 32                       // workgroups per batch in main kernel
#define ROWS_PER_WG (LMAX / SPLIT)     // 512
#define RT    32                       // feature rows per iteration
#define ITERS (ROWS_PER_WG / RT)       // 16
#define NCOL  128                      // (ql,h,kl) expanded columns
#define NJ    32                       // (ql,h) query combos per batch

// ---------------------------------------------------------------------------
// LDS layout for the main kernel (dynamic shared, ~260KB of the 320KB WGP LDS)
// ---------------------------------------------------------------------------
#define OFFS_P     0                   // P^T[b] bf16: 128 cols x 512 k  (131072 B)
#define OFFS_RAW   131072              // raw fp32 tile 32x512            (65536 B)
#define OFFS_FVRM  196608              // fv tile bf16 row-major 32x512   (32768 B)
#define OFFS_FVCM  229376              // fv tile bf16 col-major 512x32   (32768 B)
#define OFFS_WT    262144              // exp weights W^T[j][i] bf16 32x32 (2048 B)
#define OFFS_LVL   264192              // level idx per row, 32 ints        (128 B)
#define OFFS_PMAX  264320              // per (col,half) partial max       (1024 B)
#define OFFS_PSUM  265344              // per (col,half) partial sum       (1024 B)
#define OFFS_M     266368              // running max m[32]                 (128 B)
#define OFFS_L     266496              // running sum l[32]                 (128 B)
#define OFFS_SC    266624              // rescale factor per j              (128 B)
#define SMEM_BYTES 266752

// ---------------------------------------------------------------------------
// Workspace layout (bytes)
// ---------------------------------------------------------------------------
#define WS_Q     0u                    // q pre-rope, 32x512 f32            (65536)
#define WS_P     65536u                // P^T bf16, 8 x 128 x 512         (1048576)
#define WS_PACC  1114112u              // partial acc, 256wg x 32j x 512e (16777216)
#define WS_PM    17891328u             // partial m, 256x32 f32             (32768)
#define WS_PL    17924096u             // partial l, 256x32 f32             (32768)
#define WS_O     17956864u             // attention output o, 32x512 f32    (65536)

// ---------------------------------------------------------------------------
// Helpers
// ---------------------------------------------------------------------------
__device__ inline v8f wmma_bf16(v16bf a, v16bf b, v8f c) {
  return __builtin_amdgcn_wmma_f32_16x16x32_bf16(false, a, false, b, (short)0, c,
                                                 false, false);
}

// 16 bf16 operand = two 16-byte LDS loads (ds_load_b128 x2)
__device__ inline v16bf ldsv(const char* sm, int o0, int o1) {
  union { struct { u32x4 a, b; } s; v16bf v; } u;
  u.s.a = *(const u32x4*)(sm + o0);
  u.s.b = *(const u32x4*)(sm + o1);
  return u.v;
}

__device__ inline float blk_sum(float v, float* red) {
  int t = threadIdx.x;
  red[t] = v; __syncthreads();
  for (int off = 128; off > 0; off >>= 1) {
    if (t < off) red[t] += red[t + off];
    __syncthreads();
  }
  float r = red[0]; __syncthreads();
  return r;
}

#if __has_builtin(__builtin_amdgcn_tensor_load_to_lds)
#define HAVE_TDM 1
// Build a TDM descriptor (D#) per CDNA5 ISA and issue
// TENSOR_LOAD_TO_LDS (6-arg form: 4 SGPR groups + spare group + cpol).
// 2D tile (tile1==0 -> 1D). dszEnc: 0=1B,1=2B,2=4B,3=8B.
__device__ inline void tdm_load(unsigned lds_addr, const void* gp,
                                unsigned dim0, unsigned dim1,
                                unsigned tile0, unsigned tile1,
                                unsigned stride0, unsigned dszEnc) {
  unsigned long long ga = (unsigned long long)(size_t)gp;
  u32x4 g0 = { 1u,                                 // count=1, user mode
               lds_addr,
               (unsigned)ga,
               (unsigned)((ga >> 32) & 0x01FFFFFFull) | 0x80000000u }; // type=2
  i32x8 g1 = { (int)(dszEnc << 16),                                    // data_size
               (int)((dim0 & 0xFFFFu) << 16),                          // dim0 lo
               (int)(((dim0 >> 16) & 0xFFFFu) | ((dim1 & 0xFFFFu) << 16)),
               (int)(((dim1 >> 16) & 0xFFFFu) | ((tile0 & 0xFFFFu) << 16)),
               (int)(tile1 & 0xFFFFu),                                  // tile1
               (int)stride0,                                            // s0 lo32
               0, 0 };
  i32x4 gz4 = {0, 0, 0, 0};
  i32x8 gz8 = {0, 0, 0, 0, 0, 0, 0, 0};
  __builtin_amdgcn_tensor_load_to_lds(g0, g1, gz4, gz4, gz8, 0);
}
#else
#define HAVE_TDM 0
#endif

__device__ inline void tensor_wait0() {
#if HAVE_TDM
#if __has_builtin(__builtin_amdgcn_s_wait_tensorcnt)
  __builtin_amdgcn_s_wait_tensorcnt(0);
#else
  asm volatile("s_wait_tensorcnt 0x0" ::: "memory");
#endif
#endif
}

// ---------------------------------------------------------------------------
// K1: x = LN(bg); q = x @ Wq   (32 rows, pre-RoPE)
// ---------------------------------------------------------------------------
__global__ __launch_bounds__(256) void k_qproj(const float* __restrict__ bg,
                                               const float* __restrict__ Wq,
                                               const float* __restrict__ g,
                                               const float* __restrict__ bta,
                                               float* __restrict__ q) {
  __shared__ float x[EDIM];
  __shared__ float red[256];
  const int row = blockIdx.x;
  const float* r = bg + row * EDIM;
  float s = 0.f;
  for (int e = threadIdx.x; e < EDIM; e += 256) s += r[e];
  float mu = blk_sum(s, red) * (1.f / EDIM);
  float v = 0.f;
  for (int e = threadIdx.x; e < EDIM; e += 256) { float d = r[e] - mu; v += d * d; }
  float var = blk_sum(v, red) * (1.f / EDIM);
  float rstd = rsqrtf(var + 1e-5f);
  for (int e = threadIdx.x; e < EDIM; e += 256)
    x[e] = (r[e] - mu) * rstd * g[e] + bta[e];
  __syncthreads();
  for (int c = threadIdx.x; c < EDIM; c += 256) {
    float acc = 0.f;
    for (int k = 0; k < EDIM; ++k) acc += x[k] * Wq[k * EDIM + c];
    q[row * EDIM + c] = acc;
  }
}

// ---------------------------------------------------------------------------
// K2: P^T[b][col][e] = (1/8) * Wk[e, h*64:] . rope(q_head, posq - kl)  (bf16)
//     col = (ql*8 + h)*4 + kl
// ---------------------------------------------------------------------------
__global__ __launch_bounds__(256) void k_buildP(const float* __restrict__ q,
                                                const float* __restrict__ Wkv,
                                                const float* __restrict__ freqs,
                                                __bf16* __restrict__ P) {
  const int blk = blockIdx.x;                // 0..1023
  const int b = blk >> 7, col = blk & 127;
  const int ql = col >> 5, h = (col >> 2) & 7, kl = col & 3;
  __shared__ float qt[DHEAD];
  if (threadIdx.x < 32) {
    int t = threadIdx.x;
    float fr = freqs[h * 32 + t];
    int posq = (b * NLVL + ql) >> 3;         // reference: q_lvl = n // bsz
    float ang = (float)(posq - kl) * fr;
    float c = cosf(ang), sn = sinf(ang);
    const float* qh = q + (b * NLVL + ql) * EDIM + h * DHEAD;
    float x1 = qh[2 * t], x2 = qh[2 * t + 1];
    qt[2 * t]     = x1 * c - x2 * sn;
    qt[2 * t + 1] = x1 * sn + x2 * c;
  }
  __syncthreads();
  for (int e = threadIdx.x; e < EDIM; e += 256) {
    const float* w = Wkv + e * (2 * EDIM) + h * DHEAD;   // K-projection slice
    float acc = 0.f;
    for (int d = 0; d < DHEAD; ++d) acc += w[d] * qt[d];
    P[((size_t)(b * NCOL + col)) * EDIM + e] = (__bf16)(acc * 0.125f);
  }
}

// ---------------------------------------------------------------------------
// K3: main streaming kernel. 256 WGs x 256 threads (8 waves).
// Flash-style online softmax over fv rows; WMMA bf16 for scores + weighted
// accumulation; TDM async tile staging into LDS.
// ---------------------------------------------------------------------------
__global__ __launch_bounds__(256) void k_attn(const float* __restrict__ fv,
                                              const int* __restrict__ lvl,
                                              const __bf16* __restrict__ Pg,
                                              float* __restrict__ pAcc,
                                              float* __restrict__ pM,
                                              float* __restrict__ pL) {
  extern __shared__ char sm[];
  const int tid  = threadIdx.x;
  const int wave = tid >> 5;
  const int lane = tid & 31;
  const int half = lane >> 4;
  const int l16  = lane & 15;
  const int wg   = blockIdx.x;
  const int b    = wg / SPLIT;
  const int s    = wg % SPLIT;
  const long long row0 = (long long)b * LMAX + (long long)s * ROWS_PER_WG;

  float* mArr = (float*)(sm + OFFS_M);
  float* lArr = (float*)(sm + OFFS_L);
  float* scA  = (float*)(sm + OFFS_SC);
  float* pmax = (float*)(sm + OFFS_PMAX);
  float* psum = (float*)(sm + OFFS_PSUM);
  int*   lvlS = (int*)(sm + OFFS_LVL);
  const unsigned smb = (unsigned)(unsigned long long)(size_t)sm;

  // ---- preload: P^T[b] (2D tile 512x128 bf16) + first raw fv tile (1D) ----
#if HAVE_TDM
  if (tid < 32) {
    tdm_load(smb + OFFS_P,   Pg + (size_t)b * NCOL * EDIM, EDIM, NCOL, EDIM, NCOL,
             EDIM, 1u);
    tdm_load(smb + OFFS_RAW, fv + row0 * EDIM, RT * EDIM, 0, RT * EDIM, 0, 0, 2u);
  }
#else
  {
    const u32x4* srcp = (const u32x4*)(Pg + (size_t)b * NCOL * EDIM);
    u32x4* dstp = (u32x4*)(sm + OFFS_P);
    for (int t = tid; t < NCOL * EDIM * 2 / 16; t += 256) dstp[t] = srcp[t];
    const f32x4* srcr = (const f32x4*)(fv + row0 * EDIM);
    f32x4* dstr = (f32x4*)(sm + OFFS_RAW);
    for (int t = tid; t < RT * EDIM / 4; t += 256) dstr[t] = srcr[t];
  }
#endif
  if (tid < NJ) { mArr[tid] = -__builtin_inff(); lArr[tid] = 0.f; }

  v8f acc[4][2];                     // ACC^T tiles: e = (wave*4+mt)*16, j = nt*16
  for (int a = 0; a < 4; ++a) for (int n = 0; n < 2; ++n) acc[a][n] = v8f{};

  if (tid < 32) tensor_wait0();
  __syncthreads();

  for (int it = 0; it < ITERS; ++it) {
    // ---- convert raw f32 -> bf16 row-major + col-major copies ----
    for (int rdx = 0; rdx < 16; ++rdx) {
      int qi = rdx * 256 + tid;                   // 0..4095 float4s
      int i  = qi >> 7;                           // row 0..31
      int e0 = (qi & 127) * 4;                    // col base
      f32x4 v = ((const f32x4*)(sm + OFFS_RAW))[qi];
      union { __bf16 h[4]; unsigned long long u; } pk;
      pk.h[0] = (__bf16)v[0]; pk.h[1] = (__bf16)v[1];
      pk.h[2] = (__bf16)v[2]; pk.h[3] = (__bf16)v[3];
      *(unsigned long long*)(sm + OFFS_FVRM + (i * EDIM + e0) * 2) = pk.u;
      char* cm = sm + OFFS_FVCM + i * 2;
      *(__bf16*)(cm + (e0 + 0) * 64) = pk.h[0];
      *(__bf16*)(cm + (e0 + 1) * 64) = pk.h[1];
      *(__bf16*)(cm + (e0 + 2) * 64) = pk.h[2];
      *(__bf16*)(cm + (e0 + 3) * 64) = pk.h[3];
    }
    if (tid < RT) lvlS[tid] = lvl[row0 + (long long)it * RT + tid];
    __syncthreads();

    // ---- prefetch next raw tile (overlaps with all compute below) ----
    if (it + 1 < ITERS) {
#if HAVE_TDM
      if (tid < 32)
        tdm_load(smb + OFFS_RAW, fv + (row0 + (long long)(it + 1) * RT) * EDIM,
                 RT * EDIM, 0, RT * EDIM, 0, 0, 2u);
#else
      const f32x4* srcr =
          (const f32x4*)(fv + (row0 + (long long)(it + 1) * RT) * EDIM);
      f32x4* dstr = (f32x4*)(sm + OFFS_RAW);
      for (int t = tid; t < RT * EDIM / 4; t += 256) dstr[t] = srcr[t];
#endif
    }

    // ---- score GEMM: S(32 x 128) = FV(32x512) @ P^T cols; wave -> 16 cols --
    const int cbase = wave * 16;
    v8f sc0 = v8f{}, sc1 = v8f{};
    for (int kt = 0; kt < 16; ++kt) {
      int k0   = kt * 32;
      int offB = OFFS_P + ((cbase + l16) * EDIM + k0) * 2 + half * 32;
      v16bf B  = ldsv(sm, offB, offB + 16);
      int offA = OFFS_FVRM + (l16 * EDIM + k0) * 2 + half * 32;
      v16bf A0 = ldsv(sm, offA, offA + 16);
      v16bf A1 = ldsv(sm, offA + 16 * EDIM * 2, offA + 16 * EDIM * 2 + 16);
      sc0 = wmma_bf16(A0, B, sc0);
      sc1 = wmma_bf16(A1, B, sc1);
    }

    // ---- masked tile max (only elements whose kl matches the row level) ----
    const int c  = cbase + l16;
    const int j  = c >> 2;
    const int kl = c & 3;
    int lv[16];
#pragma unroll
    for (int g = 0; g < 8; ++g) {
      lv[g]     = lvlS[g + 8 * half];
      lv[8 + g] = lvlS[16 + g + 8 * half];
    }
    float lm = -__builtin_inff();
#pragma unroll
    for (int g = 0; g < 8; ++g) {
      if (lv[g]     == kl) lm = fmaxf(lm, sc0[g]);
      if (lv[8 + g] == kl) lm = fmaxf(lm, sc1[g]);
    }
    pmax[c * 2 + half] = lm;
    __syncthreads();

    if (tid < NJ) {
      float tm = -__builtin_inff();
#pragma unroll
      for (int u = 0; u < 8; ++u)
        tm = fmaxf(tm, pmax[(tid * 4 + (u >> 1)) * 2 + (u & 1)]);
      float mo = mArr[tid];
      float mn = fmaxf(mo, tm);
      scA[tid]  = __expf(mo - mn);
      mArr[tid] = mn;
    }
    __syncthreads();

    // ---- rescale accumulators; masked exp weights -> W^T[j][i] in LDS ----
    float s0 = scA[l16], s1 = scA[16 + l16];
#pragma unroll
    for (int mt = 0; mt < 4; ++mt)
#pragma unroll
      for (int g = 0; g < 8; ++g) { acc[mt][0][g] *= s0; acc[mt][1][g] *= s1; }

    float mnJ  = mArr[j];
    float lsum = 0.f;
#pragma unroll
    for (int g = 0; g < 8; ++g) {
      int i0 = g + 8 * half;
      if (lv[g] == kl) {
        float e = __expf(sc0[g] - mnJ);
        lsum += e;
        *(__bf16*)(sm + OFFS_WT + j * 64 + i0 * 2) = (__bf16)e;
      }
      int i1 = 16 + g + 8 * half;
      if (lv[8 + g] == kl) {
        float e = __expf(sc1[g] - mnJ);
        lsum += e;
        *(__bf16*)(sm + OFFS_WT + j * 64 + i1 * 2) = (__bf16)e;
      }
    }
    psum[c * 2 + half] = lsum;
    __syncthreads();

    if (tid < NJ) {
      float add = 0.f;
#pragma unroll
      for (int u = 0; u < 8; ++u)
        add += psum[(tid * 4 + (u >> 1)) * 2 + (u & 1)];
      lArr[tid] = lArr[tid] * scA[tid] + add;
    }

    // ---- accumulation GEMM: ACC^T(512x32) += FV^T(512x32) @ W(32x32) ----
    int offB0 = OFFS_WT + l16 * 64 + half * 32;
    v16bf Bw0 = ldsv(sm, offB0, offB0 + 16);
    int offB1 = offB0 + 16 * 64;
    v16bf Bw1 = ldsv(sm, offB1, offB1 + 16);
#pragma unroll
    for (int mt = 0; mt < 4; ++mt) {
      int e0   = (wave * 4 + mt) * 16;
      int offA = OFFS_FVCM + (e0 + l16) * 64 + half * 32;
      v16bf A  = ldsv(sm, offA, offA + 16);
      acc[mt][0] = wmma_bf16(A, Bw0, acc[mt][0]);
      acc[mt][1] = wmma_bf16(A, Bw1, acc[mt][1]);
    }

    if (tid < 32) tensor_wait0();   // next raw tile arrived
    __syncthreads();                // everyone done with FVRM/FVCM/WT
  }

  // ---- write partials ----
#pragma unroll
  for (int mt = 0; mt < 4; ++mt)
#pragma unroll
    for (int nt = 0; nt < 2; ++nt)
#pragma unroll
      for (int g = 0; g < 8; ++g) {
        int e  = (wave * 4 + mt) * 16 + g + 8 * half;
        int jj = nt * 16 + l16;
        pAcc[((size_t)wg * NJ + jj) * EDIM + e] = acc[mt][nt][g];
      }
  if (tid < NJ) {
    pM[wg * NJ + tid] = mArr[tid];
    pL[wg * NJ + tid] = lArr[tid];
  }
}

// ---------------------------------------------------------------------------
// K4: merge SPLIT partials per (b,j); fold Wv: o_head = (sbar @ Wv_head) / l*
// ---------------------------------------------------------------------------
__global__ __launch_bounds__(256) void k_reduce(const float* __restrict__ pAcc,
                                                const float* __restrict__ pM,
                                                const float* __restrict__ pL,
                                                const float* __restrict__ Wkv,
                                                float* __restrict__ o) {
  const int b = blockIdx.x >> 5, j = blockIdx.x & 31;
  __shared__ float sbar[EDIM];
  __shared__ float ml[SPLIT], ll[SPLIT];
  if (threadIdx.x < SPLIT) {
    int wg = b * SPLIT + threadIdx.x;
    ml[threadIdx.x] = pM[wg * NJ + j];
    ll[threadIdx.x] = pL[wg * NJ + j];
  }
  __syncthreads();
  float mstar = -__builtin_inff();
  for (int t = 0; t < SPLIT; ++t) mstar = fmaxf(mstar, ml[t]);
  float lstar = 0.f;
  for (int t = 0; t < SPLIT; ++t) lstar += __expf(ml[t] - mstar) * ll[t];
  for (int e = threadIdx.x; e < EDIM; e += 256) {
    float a = 0.f;
    for (int t = 0; t < SPLIT; ++t)
      a += __expf(ml[t] - mstar) *
           pAcc[((size_t)(b * SPLIT + t) * NJ + j) * EDIM + e];
    sbar[e] = a;
  }
  __syncthreads();
  const int ql = j >> 3, h = j & 7;
  for (int d = threadIdx.x; d < DHEAD; d += 256) {
    float a = 0.f;
    for (int e = 0; e < EDIM; ++e)
      a += sbar[e] * Wkv[e * (2 * EDIM) + EDIM + h * DHEAD + d];
    o[(b * NLVL + ql) * EDIM + h * DHEAD + d] = a / lstar;
  }
}

// ---------------------------------------------------------------------------
// K5: r = o@Wo + bg; y = LN(r); out = r + relu(y@W1+b1)@W2 + b2
// ---------------------------------------------------------------------------
__global__ __launch_bounds__(256) void k_epilogue(const float* __restrict__ o,
                                                  const float* __restrict__ bg,
                                                  const float* __restrict__ Wo,
                                                  const float* __restrict__ lfg,
                                                  const float* __restrict__ lfb,
                                                  const float* __restrict__ W1,
                                                  const float* __restrict__ b1,
                                                  const float* __restrict__ W2,
                                                  const float* __restrict__ b2,
                                                  float* __restrict__ out) {
  __shared__ float r[EDIM];
  __shared__ float xln[EDIM];
  __shared__ float h1[FFND];
  __shared__ float red[256];
  const int row = blockIdx.x;
  const float* orow = o + row * EDIM;
  for (int e = threadIdx.x; e < EDIM; e += 256) {
    float a = 0.f;
    for (int k = 0; k < EDIM; ++k) a += orow[k] * Wo[k * EDIM + e];
    r[e] = a + bg[row * EDIM + e];
  }
  __syncthreads();
  float s = 0.f;
  for (int e = threadIdx.x; e < EDIM; e += 256) s += r[e];
  float mu = blk_sum(s, red) * (1.f / EDIM);
  float v = 0.f;
  for (int e = threadIdx.x; e < EDIM; e += 256) { float d = r[e] - mu; v += d * d; }
  float var  = blk_sum(v, red) * (1.f / EDIM);
  float rstd = rsqrtf(var + 1e-5f);
  for (int e = threadIdx.x; e < EDIM; e += 256)
    xln[e] = (r[e] - mu) * rstd * lfg[e] + lfb[e];
  __syncthreads();
  for (int f = threadIdx.x; f < FFND; f += 256) {
    float a = b1[f];
    for (int e = 0; e < EDIM; ++e) a += xln[e] * W1[e * FFND + f];
    h1[f] = fmaxf(a, 0.f);
  }
  __syncthreads();
  for (int e = threadIdx.x; e < EDIM; e += 256) {
    float a = b2[e];
    for (int f = 0; f < FFND; ++f) a += h1[f] * W2[f * EDIM + e];
    out[row * EDIM + e] = r[e] + a;
  }
}

// ---------------------------------------------------------------------------
// Launch
// ---------------------------------------------------------------------------
extern "C" void kernel_launch(void* const* d_in, const int* in_sizes, int n_in,
                              void* d_out, int out_size, void* d_ws, size_t ws_size,
                              hipStream_t stream) {
  const float* bg    = (const float*)d_in[0];
  const float* fv    = (const float*)d_in[1];
  const int*   flv   = (const int*)d_in[3];
  const float* Wq    = (const float*)d_in[4];
  const float* Wkv   = (const float*)d_in[5];
  const float* Wo    = (const float*)d_in[6];
  const float* rfreq = (const float*)d_in[7];
  const float* lag   = (const float*)d_in[8];
  const float* lab   = (const float*)d_in[9];
  const float* lfg   = (const float*)d_in[10];
  const float* lfb   = (const float*)d_in[11];
  const float* W1    = (const float*)d_in[12];
  const float* b1    = (const float*)d_in[13];
  const float* W2    = (const float*)d_in[14];
  const float* b2    = (const float*)d_in[15];
  float* out = (float*)d_out;

  char* ws = (char*)d_ws;
  float*  q    = (float*)(ws + WS_Q);
  __bf16* P    = (__bf16*)(ws + WS_P);
  float*  pAcc = (float*)(ws + WS_PACC);
  float*  pM   = (float*)(ws + WS_PM);
  float*  pL   = (float*)(ws + WS_PL);
  float*  o    = (float*)(ws + WS_O);

  (void)in_sizes; (void)n_in; (void)out_size; (void)ws_size;

  (void)hipFuncSetAttribute((const void*)k_attn,
                            hipFuncAttributeMaxDynamicSharedMemorySize,
                            SMEM_BYTES);

  k_qproj   <<<BSZ * NLVL, 256, 0, stream>>>(bg, Wq, lag, lab, q);
  k_buildP  <<<BSZ * NCOL, 256, 0, stream>>>(q, Wkv, rfreq, P);
  k_attn    <<<BSZ * SPLIT, 256, SMEM_BYTES, stream>>>(fv, flv, P, pAcc, pM, pL);
  k_reduce  <<<BSZ * NJ, 256, 0, stream>>>(pAcc, pM, pL, Wkv, o);
  k_epilogue<<<BSZ * NLVL, 256, 0, stream>>>(o, bg, Wo, lfg, lfb, W1, b1, W2, b2,
                                             out);
}
